// SNN_74723841016014
// MI455X (gfx1250) — matrix operand order repaired and checked by reading, not compiled, not run on hardware
//
#include <hip/hip_runtime.h>
#include <stdint.h>

typedef __attribute__((ext_vector_type(16))) _Float16 v16h;
typedef __attribute__((ext_vector_type(8)))  _Float16 v8h;
typedef __attribute__((ext_vector_type(8)))  float    v8f;
typedef __attribute__((ext_vector_type(4)))  float    v4f;

namespace snn {
constexpr int T_STEPS = 128;
constexpr int BATCH   = 256;
constexpr int IN_DIM  = 1024;
constexpr int H1_DIM  = 2048;
constexpr int H2_DIM  = 2048;
constexpr int OUT_DIM = 512;

constexpr float DT_C     = 0.01f;
constexpr float MIN_TAU  = 0.1f;
constexpr float THRESH   = 1.0f;

constexpr int BM   = 128;  // batch rows per block
constexpr int BN   = 128;  // neurons per block
constexpr int KC   = 32;   // K chunk (one f16 WMMA)
constexpr int KP   = 40;   // padded LDS row stride (halves) vs bank conflicts
constexpr int NTHR = 256;  // 8 waves (wave32)
}

// ---------------------------------------------------------------------------
// Fused (spikes @ W^T + b) GEMM + LIF, double-buffered A staging.
//   A    : (BATCH, K) spikes; fp32 for layer 1, f16 for layers 2/3
//   W    : (N, K) f16 weights (row-major, K contiguous, L2-resident)
//   V    : (BATCH, N) fp32 persistent membrane (RMW across timesteps)
//   Sout : (BATCH, N) spikes out; f16 (hidden) or fp32 (last layer)
//   Vtmp : (BATCH, N) fp32 pre-reset membrane (last layer only)
// Block = 128x128 tile, 8 waves, each wave 2 M-tiles x 4 N-tiles
// -> 8 v_wmma_f32_16x16x32_f16 per K-chunk per wave.
// f16 A tiles are staged with CDNA5 async-to-LDS DMA (ASYNCcnt) so the copy
// of chunk k+1 overlaps the WMMAs of chunk k; one barrier per chunk.
// ---------------------------------------------------------------------------
template <bool A_IS_F16, bool LAST, int N, int K>
__global__ __launch_bounds__(snn::NTHR)
void snn_gemm_lif_kernel(const void* __restrict__ Ain,
                         const _Float16* __restrict__ W,
                         const float* __restrict__ bias,
                         const float* __restrict__ alpha,
                         float* __restrict__ V,
                         void* __restrict__ SoutV,
                         float* __restrict__ Vtmp)
{
    using namespace snn;
    __shared__ _Float16 As[2][BM * KP];       // 2 x 10 KB of 320 KB WGP LDS

    const int tid  = threadIdx.x;
    const int lane = tid & 31;
    const int wave = tid >> 5;
    const int wm   = (wave >> 1) * 32;        // 4 wave-rows  (2 M-tiles each)
    const int wn   = (wave & 1)  * 64;        // 2 wave-cols  (4 N-tiles each)

    const int bm = blockIdx.x * BM;
    const int bn = blockIdx.y * BN;

    const int m    = lane & 15;               // M (A) / N (B) index in tile
    const int half = lane >> 4;               // lane group -> K sub-range

    // ---- A-tile staging helper (one K-chunk into LDS buffer `buf`) ----
    auto stage = [&](int k0, int buf) {
        if constexpr (A_IS_F16) {
            // CDNA5 async DMA: f16 spikes -> LDS, no VGPR round-trip.
            // Tile = 128 rows x 64 B = 512 x 16 B chunks; 2 per thread.
            const _Float16* Af = (const _Float16*)Ain;
#pragma unroll
            for (int j = 0; j < 2; ++j) {
                int row = (tid + j * NTHR) >> 2;   // 0..127
                int c   = (tid + j * NTHR) & 3;    // 16B chunk within row
                uint32_t ldsoff =
                    (uint32_t)(uintptr_t)&As[buf][row * KP + c * 8];
                const _Float16* g = Af + (bm + row) * K + k0 + c * 8;
                asm volatile(
                    "global_load_async_to_lds_b128 %0, %1, off"
                    :: "v"(ldsoff), "v"(g) : "memory");
            }
        } else {
            // Layer 1: fp32 spikes, convert while staging. 1024 float4.
            const float* Af = (const float*)Ain;
#pragma unroll
            for (int j = 0; j < 4; ++j) {
                int row = (tid + j * NTHR) >> 3;        // 0..127
                int c4  = ((tid + j * NTHR) & 7) << 2;  // 0,4,...,28
                v4f x = *(const v4f*)(Af + (bm + row) * K + k0 + c4);
                _Float16* dst = &As[buf][row * KP + c4];
                dst[0] = (_Float16)x.x; dst[1] = (_Float16)x.y;
                dst[2] = (_Float16)x.z; dst[3] = (_Float16)x.w;
            }
        }
    };
    auto stage_fence = [&]() {
        if constexpr (A_IS_F16)
            asm volatile("s_wait_asynccnt 0x0" ::: "memory");
        // (fp32 path: __syncthreads' dscnt wait covers the ds_stores)
    };

    v8f acc[2][4];
#pragma unroll
    for (int i = 0; i < 2; ++i)
#pragma unroll
        for (int j = 0; j < 4; ++j)
            acc[i][j] = v8f{0.f, 0.f, 0.f, 0.f, 0.f, 0.f, 0.f, 0.f};

    // Prologue: stage chunk 0
    stage(0, 0);
    stage_fence();
    __syncthreads();

#pragma unroll 1
    for (int k0 = 0; k0 < K; k0 += KC) {
        const int p = (k0 / KC) & 1;

        // Kick off next chunk's staging into the other buffer; the async
        // DMA runs while this chunk's WMMAs execute.
        if (k0 + KC < K) stage(k0 + KC, p ^ 1);

        // A fragments: 16x32 f16; lane (m, half) holds K run
        // [half*16, half*16+16) (contiguous, matches B)
        v16h afrag[2];
#pragma unroll
        for (int mt = 0; mt < 2; ++mt) {
            int row = wm + mt * 16 + m;
            v8h lo = *(const v8h*)&As[p][row * KP + half * 16];
            v8h hi = *(const v8h*)&As[p][row * KP + half * 16 + 8];
            afrag[mt] = __builtin_shufflevector(lo, hi,
                0, 1, 2, 3, 4, 5, 6, 7, 8, 9, 10, 11, 12, 13, 14, 15);
        }

        // B fragments: 32x16 f16 (= W^T tile); lane n = m, lane-halves split
        // K into two 16-element contiguous runs straight from global/L2.
        v16h bfrag[4];
#pragma unroll
        for (int nt = 0; nt < 4; ++nt) {
            const _Float16* wp =
                W + (bn + wn + nt * 16 + m) * K + k0 + half * 16;
            bfrag[nt] = *(const v16h*)wp;
            __builtin_prefetch(wp + KC, 0, 3);   // next K-chunk of this row
        }

#pragma unroll
        for (int mt = 0; mt < 2; ++mt)
#pragma unroll
            for (int nt = 0; nt < 4; ++nt)
                acc[mt][nt] = __builtin_amdgcn_wmma_f32_16x16x32_f16(
                    false, afrag[mt], false, bfrag[nt],
                    (short)0, acc[mt][nt], false, false);

        // One barrier per chunk: publishes buffer p^1 (staging done) and
        // retires this chunk's reads of buffer p before it is rewritten.
        stage_fence();
        __syncthreads();
    }

    // Epilogue: bias + LIF. D layout: VGPR i -> M = i + 8*half, lane%16 = N.
    // All indices fit in 31 bits -> int offsets, SGPR-base addressing.
#pragma unroll
    for (int mt = 0; mt < 2; ++mt) {
#pragma unroll
        for (int nt = 0; nt < 4; ++nt) {
            const int gn = bn + wn + nt * 16 + m;
            const float bb = bias[gn];
            const float al = alpha[gn];
#pragma unroll
            for (int i = 0; i < 8; ++i) {
                const int gm  = bm + wm + mt * 16 + half * 8 + i;
                const int idx = gm * N + gn;
                const float cur  = acc[mt][nt][i] + bb;
                const float v    = V[idx];
                const float vnew = v + al * (cur - v);      // leaky integrate
                const float spk  = (vnew > snn::THRESH) ? 1.0f : 0.0f;
                V[idx] = vnew * (1.0f - spk);               // zero reset
                if constexpr (LAST) {
                    ((float*)SoutV)[idx] = spk;
                    Vtmp[idx] = vnew;                       // pre-reset v
                } else {
                    ((_Float16*)SoutV)[idx] = (_Float16)spk; // exact: {0,1}
                }
            }
        }
    }
}

// ---------------------------------------------------------------------------
// Prep kernels
// ---------------------------------------------------------------------------
__global__ void snn_f32_to_f16_kernel(const float* __restrict__ src,
                                      _Float16* __restrict__ dst, int n) {
    for (int i = blockIdx.x * blockDim.x + threadIdx.x; i < n;
         i += gridDim.x * blockDim.x)
        dst[i] = (_Float16)src[i];
}

__global__ void snn_alpha_kernel(const float* __restrict__ w,
                                 float* __restrict__ a, int n) {
    for (int i = blockIdx.x * blockDim.x + threadIdx.x; i < n;
         i += gridDim.x * blockDim.x) {
        const float sig = 1.0f / (1.0f + __expf(-w[i]));
        a[i] = snn::DT_C / (snn::MIN_TAU + sig);            // DT / tau
    }
}

__global__ void snn_zero_kernel(float* __restrict__ p, int n) {
    for (int i = blockIdx.x * blockDim.x + threadIdx.x; i < n;
         i += gridDim.x * blockDim.x)
        p[i] = 0.0f;
}

// ---------------------------------------------------------------------------
extern "C" void kernel_launch(void* const* d_in, const int* in_sizes, int n_in,
                              void* d_out, int out_size, void* d_ws, size_t ws_size,
                              hipStream_t stream) {
    using namespace snn;
    const float* s   = (const float*)d_in[0];  // (T,B,IN)
    const float* W1  = (const float*)d_in[1];
    const float* b1  = (const float*)d_in[2];
    const float* wt1 = (const float*)d_in[3];
    const float* W2  = (const float*)d_in[4];
    const float* b2  = (const float*)d_in[5];
    const float* wt2 = (const float*)d_in[6];
    const float* W3  = (const float*)d_in[7];
    const float* b3  = (const float*)d_in[8];
    const float* wt3 = (const float*)d_in[9];
    float* out = (float*)d_out;

    // Workspace carve-up (~21 MB; L2-resident on MI455X's 192 MB L2)
    char* p = (char*)d_ws;
    auto take = [&](size_t bytes) -> char* {
        char* r = p;
        p += (bytes + 255) & ~(size_t)255;
        return r;
    };
    _Float16* W1h = (_Float16*)take((size_t)H1_DIM  * IN_DIM * 2);
    _Float16* W2h = (_Float16*)take((size_t)H2_DIM  * H1_DIM * 2);
    _Float16* W3h = (_Float16*)take((size_t)OUT_DIM * H2_DIM * 2);
    float* a1 = (float*)take((size_t)H1_DIM  * 4);
    float* a2 = (float*)take((size_t)H2_DIM  * 4);
    float* a3 = (float*)take((size_t)OUT_DIM * 4);
    float* v1 = (float*)take((size_t)BATCH * H1_DIM  * 4);
    float* v2 = (float*)take((size_t)BATCH * H2_DIM  * 4);
    float* v3 = (float*)take((size_t)BATCH * OUT_DIM * 4);
    _Float16* s1 = (_Float16*)take((size_t)BATCH * H1_DIM * 2);
    _Float16* s2 = (_Float16*)take((size_t)BATCH * H2_DIM * 2);

    // ---- Precompute: f16 weights, per-neuron alpha, zeroed membranes ----
    snn_f32_to_f16_kernel<<<512, 256, 0, stream>>>(W1, W1h, H1_DIM * IN_DIM);
    snn_f32_to_f16_kernel<<<512, 256, 0, stream>>>(W2, W2h, H2_DIM * H1_DIM);
    snn_f32_to_f16_kernel<<<512, 256, 0, stream>>>(W3, W3h, OUT_DIM * H2_DIM);

    snn_alpha_kernel<<<8, 256, 0, stream>>>(wt1, a1, H1_DIM);
    snn_alpha_kernel<<<8, 256, 0, stream>>>(wt2, a2, H2_DIM);
    snn_alpha_kernel<<<8, 256, 0, stream>>>(wt3, a3, OUT_DIM);

    snn_zero_kernel<<<512, 256, 0, stream>>>(v1, BATCH * H1_DIM);
    snn_zero_kernel<<<512, 256, 0, stream>>>(v2, BATCH * H2_DIM);
    snn_zero_kernel<<<512, 256, 0, stream>>>(v3, BATCH * OUT_DIM);

    // ---- Sequential scan over timesteps; stream order carries the deps ----
    const dim3 blk(NTHR);
    const dim3 g1(BATCH / BM, H1_DIM  / BN);   // (2, 16)
    const dim3 g2(BATCH / BM, H2_DIM  / BN);   // (2, 16)
    const dim3 g3(BATCH / BM, OUT_DIM / BN);   // (2, 4)

    float* out_s = out;                                         // (T,B,OUT)
    float* out_v = out + (size_t)T_STEPS * BATCH * OUT_DIM;     // (T,B,OUT)

    for (int t = 0; t < T_STEPS; ++t) {
        const float* st = s + (size_t)t * BATCH * IN_DIM;
        snn_gemm_lif_kernel<false, false, H1_DIM, IN_DIM>
            <<<g1, blk, 0, stream>>>(st, W1h, b1, a1, v1, s1, nullptr);
        snn_gemm_lif_kernel<true, false, H2_DIM, H1_DIM>
            <<<g2, blk, 0, stream>>>(s1, W2h, b2, a2, v2, s2, nullptr);
        snn_gemm_lif_kernel<true, true, OUT_DIM, H2_DIM>
            <<<g3, blk, 0, stream>>>(s2, W3h, b3, a3, v3,
                out_s + (size_t)t * BATCH * OUT_DIM,
                out_v + (size_t)t * BATCH * OUT_DIM);
    }
}
